// CachedAttention_7619271983335
// MI455X (gfx1250) — matrix operand order repaired and checked by reading, not compiled, not run on hardware
//
#include <hip/hip_runtime.h>

// ---------------- problem constants ----------------
#define BATCH   4
#define LNEW    2048
#define SCACHE  2048
#define DMODEL  2048
#define TTOT    (LNEW + SCACHE)   // 4096

#define ASGLOBAL __attribute__((address_space(1)))
#define ASLOCAL  __attribute__((address_space(3)))

typedef int i32x4 __attribute__((vector_size(16)));   // matches builtin param type

#if __has_builtin(__builtin_amdgcn_global_load_async_to_lds_b128) && \
    __has_builtin(__builtin_amdgcn_s_wait_asynccnt)
#define HAVE_ASYNC_LDS 1
#else
#define HAVE_ASYNC_LDS 0
#endif

// ---------------- WMMA fragment types ----------------
typedef __attribute__((ext_vector_type(16))) __bf16 v16bf;
typedef __attribute__((ext_vector_type(8)))  float  v8f;

union Frag16 {
  v16bf v;
  unsigned u[8];
  unsigned short s[16];
};

static __device__ __forceinline__ unsigned short f32_to_bf16(float f) {
  union { float f; unsigned u; } x; x.f = f;
  unsigned r = x.u + 0x7FFFu + ((x.u >> 16) & 1u);   // round-to-nearest-even
  return (unsigned short)(r >> 16);
}
static __device__ __forceinline__ float bf16_to_f32(unsigned short s) {
  union { unsigned u; float f; } x; x.u = ((unsigned)s) << 16;
  return x.f;
}

static __device__ __forceinline__ v8f wmma_bf16(v16bf a, v16bf b, v8f c) {
  // (neg_a, A, neg_b, B, c_mod, C, reuse_a, reuse_b)
  return __builtin_amdgcn_wmma_f32_16x16x32_bf16(false, a, false, b, (short)0, c,
                                                 false, false);
}

// A-matrix 16x32 bf16 fragment from row-major [16 x ld] at column k0.
// Contiguous k order (same as load_bt_frag): lanes 0-15 hold M=lane; VGPR j (pair) -> K = 16*(lane>>4) + 2*j
static __device__ __forceinline__ v16bf load_a_frag(const unsigned short* a,
                                                    size_t ld, int k0) {
  int lane = threadIdx.x & 31;
  int l = lane & 15, h = lane >> 4;
  const unsigned short* row = a + (size_t)l * ld + (size_t)k0;
  Frag16 f;
#pragma unroll
  for (int j = 0; j < 8; ++j) {
    int k = (h << 4) + (j << 1);
    f.u[j] = *(const unsigned*)(row + k);
  }
  return f.v;
}

// B-matrix 32x16 fragment where B = Mrows^T: element (k, n) = Mrows[n][k].
// Lane column n = lane&15, contiguous K = k0 + 16*(lane>>4) .. +15 -> two b128 loads.
static __device__ __forceinline__ v16bf load_bt_frag(const unsigned short* mrows,
                                                     size_t ld, int k0) {
  int lane = threadIdx.x & 31;
  int l = lane & 15, h = lane >> 4;
  const uint4* q = (const uint4*)(mrows + (size_t)l * ld + (size_t)k0 + (h << 4));
  Frag16 f;
  uint4 x = q[0];
  uint4 y = q[1];
  f.u[0] = x.x; f.u[1] = x.y; f.u[2] = x.z; f.u[3] = x.w;
  f.u[4] = y.x; f.u[5] = y.y; f.u[6] = y.z; f.u[7] = y.w;
  return f.v;
}

// C/D f32 16x16: lane col n = lane&15; VGPR r -> row r + 8*(lane>>4).
static __device__ __forceinline__ void store_c_bf16(unsigned short* out, size_t ld,
                                                    v8f c, float scale) {
  int lane = threadIdx.x & 31;
  int l = lane & 15, h = lane >> 4;
#pragma unroll
  for (int r = 0; r < 8; ++r)
    out[(size_t)(r + (h << 3)) * ld + l] = f32_to_bf16(c[r] * scale);
}
static __device__ __forceinline__ void store_c_f32(float* out, size_t ld, v8f c) {
  int lane = threadIdx.x & 31;
  int l = lane & 15, h = lane >> 4;
#pragma unroll
  for (int r = 0; r < 8; ++r)
    out[(size_t)(r + (h << 3)) * ld + l] = c[r];
}

// ---------------- conversion kernels ----------------
__global__ void cvt_f32_bf16(const float* __restrict__ in,
                             unsigned short* __restrict__ out, size_t n) {
  size_t i = ((size_t)blockIdx.x * blockDim.x + threadIdx.x) * 4;
  if (i + 3 >= n) return;
  float4 v = *(const float4*)(in + i);
  ushort4 o;
  o.x = f32_to_bf16(v.x); o.y = f32_to_bf16(v.y);
  o.z = f32_to_bf16(v.z); o.w = f32_to_bf16(v.w);
  *(ushort4*)(out + i) = o;
}

// cache_k [B, S, D] f32 -> bf16 into Kcal [B, T, D] rows L..T-1 (new K is PREPENDED)
__global__ void cvt_cache_k(const float* __restrict__ in,
                            unsigned short* __restrict__ outBase) {
  size_t i = ((size_t)blockIdx.x * blockDim.x + threadIdx.x) * 4;
  size_t row = i >> 11;            // / DMODEL
  size_t col = i & (DMODEL - 1);
  int b = (int)(row >> 11);        // / SCACHE
  size_t s = row & (SCACHE - 1);
  float4 v = *(const float4*)(in + i);
  ushort4 o;
  o.x = f32_to_bf16(v.x); o.y = f32_to_bf16(v.y);
  o.z = f32_to_bf16(v.z); o.w = f32_to_bf16(v.w);
  unsigned short* out = outBase + (((size_t)b * TTOT) + LNEW + s) * DMODEL + col;
  *(ushort4*)out = o;
}

// cache_v [B, S, D] f32 -> bf16 TRANSPOSED into Vt [B, D, T] columns L..T-1
__global__ void cvt_cache_vT(const float* __restrict__ in,
                             unsigned short* __restrict__ Vt) {
  size_t i = ((size_t)blockIdx.x * blockDim.x + threadIdx.x) * 4;
  size_t row = i >> 11;            // b*S + s
  size_t d = i & (DMODEL - 1);
  int b = (int)(row >> 11);
  size_t s = row & (SCACHE - 1);
  float4 v = *(const float4*)(in + i);
  unsigned short* base = Vt + (size_t)b * DMODEL * TTOT + (size_t)(LNEW + s);
  base[(d + 0) * TTOT] = f32_to_bf16(v.x);
  base[(d + 1) * TTOT] = f32_to_bf16(v.y);
  base[(d + 2) * TTOT] = f32_to_bf16(v.z);
  base[(d + 3) * TTOT] = f32_to_bf16(v.w);
}

// ---------------- QKV projection: Y = X @ W^T ----------------
// one wave per block; wave computes a 32x64 output tile (8 accumulators)
__global__ void proj_gemm(const unsigned short* __restrict__ Xb,
                          const unsigned short* __restrict__ Wb,   // 3x [D,D] packed
                          unsigned short* __restrict__ Qb,
                          unsigned short* __restrict__ Kcal,
                          unsigned short* __restrict__ Vt) {
  int m0 = blockIdx.x * 32;            // global row in [0, B*L)
  int n0 = blockIdx.y * 64;
  int z  = blockIdx.z;                 // 0=Q 1=K 2=V
  const unsigned short* A0 = Xb + (size_t)m0 * DMODEL;
  const unsigned short* A1 = A0 + (size_t)16 * DMODEL;
  const unsigned short* W = Wb + (size_t)z * DMODEL * DMODEL + (size_t)n0 * DMODEL;

  v8f acc[2][4];
#pragma unroll
  for (int g = 0; g < 2; ++g)
#pragma unroll
    for (int t = 0; t < 4; ++t)
      acc[g][t] = (v8f){0.f, 0.f, 0.f, 0.f, 0.f, 0.f, 0.f, 0.f};

  for (int k0 = 0; k0 < DMODEL; k0 += 32) {
    __builtin_prefetch(A0 + (size_t)(threadIdx.x & 15) * DMODEL + k0 + 64, 0, 1);
    v16bf a0 = load_a_frag(A0, DMODEL, k0);
    v16bf a1 = load_a_frag(A1, DMODEL, k0);
#pragma unroll
    for (int t = 0; t < 4; ++t) {
      v16bf b = load_bt_frag(W + (size_t)(t * 16) * DMODEL, DMODEL, k0);
      acc[0][t] = wmma_bf16(a0, b, acc[0][t]);
      acc[1][t] = wmma_bf16(a1, b, acc[1][t]);
    }
  }

  int bb = m0 >> 11;                        // / LNEW
  int l0 = m0 & (LNEW - 1);
  if (z == 2) {
    // store transposed into Vt [B, D, T]; per lane 8 consecutive M -> one b128 store
    int lane = threadIdx.x & 31;
    int l = lane & 15, h = lane >> 4;
#pragma unroll
    for (int g = 0; g < 2; ++g)
#pragma unroll
      for (int t = 0; t < 4; ++t) {
        union { uint4 q; unsigned short s[8]; } o;
#pragma unroll
        for (int r = 0; r < 8; ++r) o.s[r] = f32_to_bf16(acc[g][t][r]);
        unsigned short* dst = Vt +
            ((size_t)bb * DMODEL + (size_t)(n0 + t * 16 + l)) * TTOT +
            (size_t)(l0 + g * 16 + (h << 3));
        *(uint4*)dst = o.q;
      }
  } else {
    unsigned short* out;
    size_t orow;
    if (z == 0) { out = Qb; orow = (size_t)m0; }
    else        { out = Kcal; orow = (size_t)bb * TTOT + (size_t)l0; }
#pragma unroll
    for (int g = 0; g < 2; ++g)
#pragma unroll
      for (int t = 0; t < 4; ++t)
        store_c_bf16(out + (orow + g * 16) * DMODEL + n0 + t * 16, DMODEL,
                     acc[g][t], 1.0f);
  }
}

// ---------------- scores: S = Q @ Kcal^T * inv_scale ----------------
__global__ void score_gemm(const unsigned short* __restrict__ Qb,
                           const unsigned short* __restrict__ Kcal,
                           unsigned short* __restrict__ Sc) {
  int m0 = blockIdx.x * 32;            // query row within batch
  int n0 = blockIdx.y * 64;            // key index
  int b  = blockIdx.z;
  const unsigned short* A0 = Qb + ((size_t)b * LNEW + m0) * DMODEL;
  const unsigned short* A1 = A0 + (size_t)16 * DMODEL;
  const unsigned short* K = Kcal + ((size_t)b * TTOT + n0) * DMODEL;

  v8f acc[2][4];
#pragma unroll
  for (int g = 0; g < 2; ++g)
#pragma unroll
    for (int t = 0; t < 4; ++t)
      acc[g][t] = (v8f){0.f, 0.f, 0.f, 0.f, 0.f, 0.f, 0.f, 0.f};

  for (int k0 = 0; k0 < DMODEL; k0 += 32) {
    __builtin_prefetch(A0 + (size_t)(threadIdx.x & 15) * DMODEL + k0 + 64, 0, 1);
    v16bf a0 = load_a_frag(A0, DMODEL, k0);
    v16bf a1 = load_a_frag(A1, DMODEL, k0);
#pragma unroll
    for (int t = 0; t < 4; ++t) {
      v16bf bb = load_bt_frag(K + (size_t)(t * 16) * DMODEL, DMODEL, k0);
      acc[0][t] = wmma_bf16(a0, bb, acc[0][t]);
      acc[1][t] = wmma_bf16(a1, bb, acc[1][t]);
    }
  }

  const float inv_scale = 0.02209708691207961f;    // 1/sqrt(2048)
  unsigned short* out = Sc + ((size_t)b * LNEW + m0) * TTOT + n0;
#pragma unroll
  for (int g = 0; g < 2; ++g)
#pragma unroll
    for (int t = 0; t < 4; ++t)
      store_c_bf16(out + (size_t)(g * 16) * TTOT + t * 16, TTOT,
                   acc[g][t], inv_scale);
}

// ---------------- in-place row softmax over T=4096 bf16 ----------------
// Stages the row through LDS with CDNA5 async global->LDS copies.
__global__ void softmax_rows(unsigned short* __restrict__ Sc) {
  __shared__ unsigned short buf[TTOT];   // 8 KB
  size_t row = blockIdx.x;
  unsigned short* p = Sc + row * (size_t)TTOT;
  int t = threadIdx.x;                   // 256 threads, 16 elems each

#if HAVE_ASYNC_LDS
  __builtin_amdgcn_global_load_async_to_lds_b128(
      (ASGLOBAL i32x4*)(p + t * 8), (ASLOCAL i32x4*)(buf + t * 8), 0, 0);
  __builtin_amdgcn_global_load_async_to_lds_b128(
      (ASGLOBAL i32x4*)(p + 2048 + t * 8), (ASLOCAL i32x4*)(buf + 2048 + t * 8),
      0, 0);
  __builtin_amdgcn_s_wait_asynccnt(0);
#else
  {
    uint4* dst = (uint4*)buf;
    const uint4* src = (const uint4*)p;
    dst[t] = src[t];
    dst[t + 256] = src[t + 256];
  }
#endif
  __syncthreads();

  float v[16];
  float mx = -3.0e38f;
#pragma unroll
  for (int i = 0; i < 16; ++i) {
    v[i] = bf16_to_f32(buf[t + i * 256]);
    mx = fmaxf(mx, v[i]);
  }
  __shared__ float red[256];
  red[t] = mx;
  __syncthreads();
  for (int s = 128; s > 0; s >>= 1) {
    if (t < s) red[t] = fmaxf(red[t], red[t + s]);
    __syncthreads();
  }
  mx = red[0];
  __syncthreads();
  float sum = 0.f;
#pragma unroll
  for (int i = 0; i < 16; ++i) {
    v[i] = __expf(v[i] - mx);
    sum += v[i];
  }
  red[t] = sum;
  __syncthreads();
  for (int s = 128; s > 0; s >>= 1) {
    if (t < s) red[t] += red[t + s];
    __syncthreads();
  }
  float inv = 1.0f / red[0];
#pragma unroll
  for (int i = 0; i < 16; ++i)
    p[t + i * 256] = f32_to_bf16(v[i] * inv);
}

// ---------------- output: O = P @ Vcal = P @ Vt^T ----------------
__global__ void out_gemm(const unsigned short* __restrict__ P,   // [B, L, T] bf16
                         const unsigned short* __restrict__ Vt,  // [B, D, T] bf16
                         float* __restrict__ Out) {              // [B, L, D] f32
  int m0 = blockIdx.x * 32;
  int n0 = blockIdx.y * 64;
  int b  = blockIdx.z;
  const unsigned short* A0 = P + ((size_t)b * LNEW + m0) * TTOT;
  const unsigned short* A1 = A0 + (size_t)16 * TTOT;
  const unsigned short* V = Vt + (size_t)b * DMODEL * TTOT + (size_t)n0 * TTOT;

  v8f acc[2][4];
#pragma unroll
  for (int g = 0; g < 2; ++g)
#pragma unroll
    for (int t = 0; t < 4; ++t)
      acc[g][t] = (v8f){0.f, 0.f, 0.f, 0.f, 0.f, 0.f, 0.f, 0.f};

  for (int t0 = 0; t0 < TTOT; t0 += 32) {
    __builtin_prefetch(A0 + (size_t)(threadIdx.x & 15) * TTOT + t0 + 64, 0, 1);
    v16bf a0 = load_a_frag(A0, TTOT, t0);
    v16bf a1 = load_a_frag(A1, TTOT, t0);
#pragma unroll
    for (int t = 0; t < 4; ++t) {
      v16bf bb = load_bt_frag(V + (size_t)(t * 16) * TTOT, TTOT, t0);
      acc[0][t] = wmma_bf16(a0, bb, acc[0][t]);
      acc[1][t] = wmma_bf16(a1, bb, acc[1][t]);
    }
  }

  float* out = Out + ((size_t)b * LNEW + m0) * DMODEL + n0;
#pragma unroll
  for (int g = 0; g < 2; ++g)
#pragma unroll
    for (int t = 0; t < 4; ++t)
      store_c_f32(out + (size_t)(g * 16) * DMODEL + t * 16, DMODEL, acc[g][t]);
}

// ---------------- launcher ----------------
extern "C" void kernel_launch(void* const* d_in, const int* in_sizes, int n_in,
                              void* d_out, int out_size, void* d_ws, size_t ws_size,
                              hipStream_t stream) {
  const float* x  = (const float*)d_in[0];
  const float* ck = (const float*)d_in[1];
  const float* cv = (const float*)d_in[2];
  const float* Wq = (const float*)d_in[3];
  const float* Wk = (const float*)d_in[4];
  const float* Wv = (const float*)d_in[5];
  float* out = (float*)d_out;

  const size_t nX  = (size_t)BATCH * LNEW * DMODEL;   // 16.7M elems
  const size_t nW  = (size_t)DMODEL * DMODEL;         // 4.2M
  const size_t nKV = (size_t)BATCH * TTOT * DMODEL;   // 33.5M
  const size_t nC  = (size_t)BATCH * SCACHE * DMODEL; // 16.7M

  unsigned short* ws   = (unsigned short*)d_ws;
  unsigned short* Xb   = ws;                 // 32 MB
  unsigned short* Qb   = Xb + nX;            // 32 MB
  unsigned short* Wb   = Qb + nX;            // 24 MB (Wq|Wk|Wv)
  unsigned short* Kcal = Wb + 3 * nW;        // 64 MB  [B, T, D]
  unsigned short* Vt   = Kcal + nKV;         // 64 MB  [B, D, T] (transposed V_cal)
  unsigned short* Sc   = Vt + nKV;           // 64 MB (scores -> probs in place)

  cvt_f32_bf16<<<dim3((unsigned)(nX / 1024)), 256, 0, stream>>>(x, Xb, nX);
  cvt_f32_bf16<<<dim3((unsigned)(nW / 1024)), 256, 0, stream>>>(Wq, Wb, nW);
  cvt_f32_bf16<<<dim3((unsigned)(nW / 1024)), 256, 0, stream>>>(Wk, Wb + nW, nW);
  cvt_f32_bf16<<<dim3((unsigned)(nW / 1024)), 256, 0, stream>>>(Wv, Wb + 2 * nW, nW);
  cvt_cache_k<<<dim3((unsigned)(nC / 1024)), 256, 0, stream>>>(ck, Kcal);
  cvt_cache_vT<<<dim3((unsigned)(nC / 1024)), 256, 0, stream>>>(cv, Vt);

  proj_gemm<<<dim3((BATCH * LNEW) / 32, DMODEL / 64, 3), 32, 0, stream>>>(
      Xb, Wb, Qb, Kcal, Vt);
  score_gemm<<<dim3(LNEW / 32, TTOT / 64, BATCH), 32, 0, stream>>>(Qb, Kcal, Sc);
  softmax_rows<<<dim3(BATCH * LNEW), 256, 0, stream>>>(Sc);
  out_gemm<<<dim3(LNEW / 32, DMODEL / 64, BATCH), 32, 0, stream>>>(Sc, Vt, out);
}
